// EncoderLayer_42185168782047
// MI455X (gfx1250) — matrix operand order repaired and checked by reading, not compiled, not run on hardware
//
#include <hip/hip_runtime.h>
#include <hip/hip_bf16.h>

// ---------------------------------------------------------------------------
// Transformer encoder layer for MI455X (gfx1250, wave32, WMMA).
// Compute-bound (~140 GFLOP vs ~67MB traffic -> AI ~2000 FLOP/B), so all
// GEMMs run on v_wmma_f32_16x16x32_f16 with f32 accumulation.
// ---------------------------------------------------------------------------

typedef __attribute__((ext_vector_type(16))) _Float16 v16h;
typedef __attribute__((ext_vector_type(8)))  _Float16 v8h;
typedef __attribute__((ext_vector_type(8)))  float    v8f;

#define DMODEL 1024
#define DFF    4096
#define NH     16
#define DK     64
#define BB     2
#define LL     2048
#define BL     (BB * LL)     // 4096 rows

__device__ __forceinline__ v8f zero8() {
    v8f z = {0.f, 0.f, 0.f, 0.f, 0.f, 0.f, 0.f, 0.f};
    return z;
}

// Assemble a 16-element f16 fragment from two 8-element chunks.
__device__ __forceinline__ v16h ld_frag(const _Float16* p0, const _Float16* p1) {
    v8h a = *(const v8h*)p0;
    v8h b = *(const v8h*)p1;
    return __builtin_shufflevector(a, b, 0,1,2,3,4,5,6,7,8,9,10,11,12,13,14,15);
}

__device__ __forceinline__ v8f wmma16(v16h a, v16h b, v8f c) {
    // (neg_a, A, neg_b, B, c_mod, C, reuse_a, reuse_b)
    return __builtin_amdgcn_wmma_f32_16x16x32_f16(false, a, false, b, (short)0, c, false, false);
}

// ---------------------------------------------------------------------------
// f32 -> f16 flat convert
// ---------------------------------------------------------------------------
__global__ void cvt_f32_f16(const float* __restrict__ in, _Float16* __restrict__ out, int n) {
    for (int i = blockIdx.x * blockDim.x + threadIdx.x; i < n; i += gridDim.x * blockDim.x)
        out[i] = (_Float16)in[i];
}

// Re-layout per-head weights [H, D, 64] -> [D, H*64] (f16)
__global__ void pack_headW(const float* __restrict__ in, _Float16* __restrict__ out) {
    const int N = NH * DMODEL * DK;
    for (int i = blockIdx.x * blockDim.x + threadIdx.x; i < N; i += gridDim.x * blockDim.x) {
        int h = i >> 16;            // / (1024*64)
        int d = (i >> 6) & 1023;
        int k = i & 63;
        out[(d << 10) + (h << 6) + k] = (_Float16)in[i];
    }
}

// ---------------------------------------------------------------------------
// Tiled WMMA GEMM: C[M,N] = A[M,K](f16) x B[K,N](f16) (+bias)(+relu)
// Block tile 128x128, 8 waves, wave tile 64x32 (4x2 wmma tiles), K step 32.
// Dims used here are always multiples of the tile sizes -> no bounds checks.
// ---------------------------------------------------------------------------
__global__ __launch_bounds__(256)
void gemm128_wmma(const _Float16* __restrict__ A, int lda,
                  const _Float16* __restrict__ B, int ldb,
                  float* __restrict__ Cf, _Float16* __restrict__ Ch, int ldc,
                  const float* __restrict__ bias, int relu, int K) {
    __shared__ __align__(16) _Float16 As[128 * 40];   // row-major, +8 pad
    __shared__ __align__(16) _Float16 Bs[128 * 40];   // n-major (transposed)

    const int t    = threadIdx.x;
    const int m0   = blockIdx.y * 128;
    const int n0   = blockIdx.x * 128;
    const int wave = t >> 5, lane = t & 31;
    const int wm   = wave >> 2, wn = wave & 3;
    const int lm   = lane & 15, lg = lane >> 4;

    v8f acc[4][2];
#pragma unroll
    for (int i = 0; i < 4; ++i)
#pragma unroll
        for (int j = 0; j < 2; ++j) acc[i][j] = zero8();

    const int arow = t >> 1, aseg = (t & 1) * 16;
    const int bkr  = t >> 3, bns  = (t & 7) * 16;

    for (int kt = 0; kt < K; kt += 32) {
        // ---- stage A tile (128x32) row-major
        const _Float16* ap = A + (size_t)(m0 + arow) * lda + kt + aseg;
        *(v8h*)(As + arow * 40 + aseg)     = *(const v8h*)ap;
        *(v8h*)(As + arow * 40 + aseg + 8) = *(const v8h*)(ap + 8);
        // ---- stage B tile (32x128) transposed into n-major
        const _Float16* bp = B + (size_t)(kt + bkr) * ldb + n0 + bns;
        v8h b0 = *(const v8h*)bp;
        v8h b1 = *(const v8h*)(bp + 8);
#pragma unroll
        for (int j = 0; j < 8; ++j) Bs[(bns + j) * 40 + bkr] = b0[j];
#pragma unroll
        for (int j = 0; j < 8; ++j) Bs[(bns + 8 + j) * 40 + bkr] = b1[j];
        if (kt + 32 < K) {
            __builtin_prefetch(ap + 32, 0, 3);                       // global_prefetch_b8
            __builtin_prefetch(bp + (size_t)32 * ldb, 0, 3);
        }
        __syncthreads();

        // ---- A fragments: lane lm = row, contiguous K run of 16 at lg*16
        v16h af[4];
#pragma unroll
        for (int i = 0; i < 4; ++i) {
            const _Float16* p = As + (wm * 64 + i * 16 + lm) * 40 + lg * 16;
            af[i] = ld_frag(p, p + 8);
        }
        // ---- B fragments: lane lm = col, contiguous K run of 16 at lg*16
        v16h bf[2];
#pragma unroll
        for (int j = 0; j < 2; ++j) {
            const _Float16* p = Bs + (wn * 32 + j * 16 + lm) * 40 + lg * 16;
            bf[j] = ld_frag(p, p + 8);
        }
#pragma unroll
        for (int i = 0; i < 4; ++i)
#pragma unroll
            for (int j = 0; j < 2; ++j)
                acc[i][j] = wmma16(af[i], bf[j], acc[i][j]);
        __syncthreads();
    }

    // ---- epilogue: C layout lane lm = col, VGPR r -> row r + 8*lg
#pragma unroll
    for (int i = 0; i < 4; ++i)
#pragma unroll
        for (int j = 0; j < 2; ++j) {
            const int col  = n0 + wn * 32 + j * 16 + lm;
            const float bb = bias ? bias[col] : 0.0f;
            const int rowb = m0 + wm * 64 + i * 16 + lg * 8;
#pragma unroll
            for (int r = 0; r < 8; ++r) {
                float v = acc[i][j][r] + bb;
                if (relu) v = fmaxf(v, 0.0f);
                const size_t idx = (size_t)(rowb + r) * ldc + col;
                if (Ch) Ch[idx] = (_Float16)v;
                else    Cf[idx] = v;
            }
        }
}

// ---------------------------------------------------------------------------
// Flash attention: one wave handles a 16-row query block of one (b,h).
// 8 waves / block, 512 blocks. Online softmax over 32-key chunks.
// ---------------------------------------------------------------------------
__global__ __launch_bounds__(256)
void flash_attn(const _Float16* __restrict__ Q, const _Float16* __restrict__ Km,
                const _Float16* __restrict__ V, _Float16* __restrict__ O) {
    __shared__ __align__(16) _Float16 PsAll[8 * 16 * 40];   // P staging per wave
    __shared__ __align__(16) _Float16 VtAll[8 * 64 * 40];   // V^T staging per wave

    const int t = threadIdx.x, wave = t >> 5, lane = t & 31;
    _Float16* Ps = PsAll + wave * 16 * 40;
    _Float16* Vt = VtAll + wave * 64 * 40;

    const int gw = blockIdx.x * 8 + wave;
    const int bh = gw >> 7, qb = gw & 127;
    const int b  = bh >> 4, h = bh & 15;
    const int lm = lane & 15, lg = lane >> 4;

    const _Float16* Qp = Q  + ((size_t)(b * LL + qb * 16)) * DMODEL + h * DK;
    const _Float16* Kp = Km + ((size_t)(b * LL)) * DMODEL + h * DK;
    const _Float16* Vp = V  + ((size_t)(b * LL)) * DMODEL + h * DK;

    // Q A-fragments: two K32 chunks covering dk=64
    v16h aQ[2];
#pragma unroll
    for (int c = 0; c < 2; ++c) {
        const _Float16* p = Qp + (size_t)lm * DMODEL + c * 32 + lg * 16;
        aQ[c] = ld_frag(p, p + 8);
    }

    float mrow[8], lrow[8];
    v8f oacc[4];
#pragma unroll
    for (int r = 0; r < 8; ++r) { mrow[r] = -1e30f; lrow[r] = 0.f; }
#pragma unroll
    for (int tt = 0; tt < 4; ++tt) oacc[tt] = zero8();

    for (int kb = 0; kb < LL / 32; ++kb) {
        // ---- stage V chunk (32 x 64) transposed into LDS
        const _Float16* vp = Vp + (size_t)(kb * 32 + lane) * DMODEL;
#pragma unroll
        for (int i = 0; i < 8; ++i) {
            v8h vv = *(const v8h*)(vp + i * 8);
#pragma unroll
            for (int j = 0; j < 8; ++j) Vt[(i * 8 + j) * 40 + lane] = vv[j];
        }

        // ---- S = Q K^T for two 16-key tiles (B-frags straight from global:
        //      lane = key, VGPRs = contiguous dk run)
        v8f s0 = zero8(), s1 = zero8();
        {
            const int kc0 = kb * 32;
            const _Float16* p00 = Kp + (size_t)(kc0 + lm) * DMODEL + lg * 16;
            const _Float16* p01 = p00 + 32;
            s0 = wmma16(aQ[0], ld_frag(p00, p00 + 8), s0);
            s0 = wmma16(aQ[1], ld_frag(p01, p01 + 8), s0);
            const _Float16* p10 = Kp + (size_t)(kc0 + 16 + lm) * DMODEL + lg * 16;
            const _Float16* p11 = p10 + 32;
            s1 = wmma16(aQ[0], ld_frag(p10, p10 + 8), s1);
            s1 = wmma16(aQ[1], ld_frag(p11, p11 + 8), s1);
        }

        // ---- online softmax (C layout: lane=col, VGPR r = row r+8*lg)
#pragma unroll
        for (int r = 0; r < 8; ++r) {
            float a0 = s0[r] * 0.125f, a1 = s1[r] * 0.125f;
            float tm = fmaxf(a0, a1);
#pragma unroll
            for (int mk = 1; mk < 16; mk <<= 1) tm = fmaxf(tm, __shfl_xor(tm, mk, 32));
            float nm = fmaxf(mrow[r], tm);
            float p0 = __expf(a0 - nm), p1 = __expf(a1 - nm);
            float ts = p0 + p1;
#pragma unroll
            for (int mk = 1; mk < 16; mk <<= 1) ts += __shfl_xor(ts, mk, 32);
            float alpha = __expf(mrow[r] - nm);
            lrow[r] = lrow[r] * alpha + ts;
            mrow[r] = nm;
#pragma unroll
            for (int tt = 0; tt < 4; ++tt) oacc[tt][r] *= alpha;
            const int prow = r + lg * 8;
            Ps[prow * 40 + lm]      = (_Float16)p0;
            Ps[prow * 40 + 16 + lm] = (_Float16)p1;
        }
        __syncthreads();

        // ---- O += P (16x32) x V (32x64)
        const _Float16* pp = Ps + lm * 40 + lg * 16;
        v16h aP = ld_frag(pp, pp + 8);
#pragma unroll
        for (int tt = 0; tt < 4; ++tt) {
            const _Float16* vq = Vt + (tt * 16 + lm) * 40 + lg * 16;
            oacc[tt] = wmma16(aP, ld_frag(vq, vq + 8), oacc[tt]);
        }
        __syncthreads();
    }

    // ---- normalize and store (already in [b, l, h*dv] layout for Wo)
#pragma unroll
    for (int tt = 0; tt < 4; ++tt) {
        const int col = h * DK + tt * 16 + lm;
#pragma unroll
        for (int r = 0; r < 8; ++r) {
            const int row = b * LL + qb * 16 + r + lg * 8;
            O[(size_t)row * DMODEL + col] = (_Float16)(oacc[tt][r] / lrow[r]);
        }
    }
}

// ---------------------------------------------------------------------------
// Fused residual + LayerNorm: out = LN(A + R) * g + b ; one block per row.
// ---------------------------------------------------------------------------
__global__ __launch_bounds__(256)
void ln_res(const float* __restrict__ A, const float* __restrict__ R,
            const float* __restrict__ g, const float* __restrict__ be,
            float* __restrict__ oF, _Float16* __restrict__ oH) {
    __shared__ float rs[256], rss[256];
    const int row = blockIdx.x, t = threadIdx.x;
    const size_t base = (size_t)row * DMODEL + t * 4;
    float v[4], s = 0.f, ss = 0.f;
#pragma unroll
    for (int j = 0; j < 4; ++j) {
        float x = A[base + j] + R[base + j];
        v[j] = x; s += x; ss += x * x;
    }
    rs[t] = s; rss[t] = ss;
    __syncthreads();
    for (int o = 128; o > 0; o >>= 1) {
        if (t < o) { rs[t] += rs[t + o]; rss[t] += rss[t + o]; }
        __syncthreads();
    }
    const float mu  = rs[0] * (1.0f / DMODEL);
    const float var = rss[0] * (1.0f / DMODEL) - mu * mu;
    const float ri  = rsqrtf(var + 1e-5f);
#pragma unroll
    for (int j = 0; j < 4; ++j) {
        float y = (v[j] - mu) * ri * g[t * 4 + j] + be[t * 4 + j];
        oF[base + j] = y;
        if (oH) oH[base + j] = (_Float16)y;
    }
}

// ---------------------------------------------------------------------------
// Host orchestration
// ---------------------------------------------------------------------------
extern "C" void kernel_launch(void* const* d_in, const int* in_sizes, int n_in,
                              void* d_out, int out_size, void* d_ws, size_t ws_size,
                              hipStream_t stream) {
    (void)in_sizes; (void)n_in; (void)out_size; (void)ws_size;
    const float* x    = (const float*)d_in[0];
    const float* Wq   = (const float*)d_in[1];
    const float* bq   = (const float*)d_in[2];
    const float* Wk   = (const float*)d_in[3];
    const float* bk   = (const float*)d_in[4];
    const float* Wv   = (const float*)d_in[5];
    const float* bv   = (const float*)d_in[6];
    const float* Wo   = (const float*)d_in[7];
    const float* ln1g = (const float*)d_in[8];
    const float* ln1b = (const float*)d_in[9];
    const float* W1   = (const float*)d_in[10];
    const float* b1   = (const float*)d_in[11];
    const float* W2   = (const float*)d_in[12];
    const float* ln2g = (const float*)d_in[13];
    const float* ln2b = (const float*)d_in[14];
    float* out = (float*)d_out;

    char* w = (char*)d_ws;
    auto take = [&](size_t bytes) { char* p = w; w += (bytes + 255) & ~(size_t)255; return p; };
    _Float16* Xh  = (_Float16*)take((size_t)BL * DMODEL * 2);
    _Float16* WqP = (_Float16*)take((size_t)DMODEL * DMODEL * 2);
    _Float16* WkP = (_Float16*)take((size_t)DMODEL * DMODEL * 2);
    _Float16* WvP = (_Float16*)take((size_t)DMODEL * DMODEL * 2);
    _Float16* WoH = (_Float16*)take((size_t)DMODEL * DMODEL * 2);
    _Float16* W1H = (_Float16*)take((size_t)DMODEL * DFF * 2);
    _Float16* W2H = (_Float16*)take((size_t)DFF * DMODEL * 2);
    _Float16* Qh  = (_Float16*)take((size_t)BL * DMODEL * 2);
    _Float16* Kh  = (_Float16*)take((size_t)BL * DMODEL * 2);
    _Float16* Vh  = (_Float16*)take((size_t)BL * DMODEL * 2);
    _Float16* Oh  = (_Float16*)take((size_t)BL * DMODEL * 2);
    float*    mha = (float*)take((size_t)BL * DMODEL * 4);     // reused for FFN out
    float*    h1f = (float*)take((size_t)BL * DMODEL * 4);
    _Float16* h1h = (_Float16*)take((size_t)BL * DMODEL * 2);
    _Float16* ffa = (_Float16*)take((size_t)BL * DFF * 2);
    float*    ff  = mha;   // mha dead after ln1 -> reuse

    const dim3 blk(256);
    // precision conversion / weight packing
    cvt_f32_f16<<<dim3(2048), blk, 0, stream>>>(x, Xh, BL * DMODEL);
    pack_headW<<<dim3(1024), blk, 0, stream>>>(Wq, WqP);
    pack_headW<<<dim3(1024), blk, 0, stream>>>(Wk, WkP);
    pack_headW<<<dim3(1024), blk, 0, stream>>>(Wv, WvP);
    cvt_f32_f16<<<dim3(1024), blk, 0, stream>>>(Wo, WoH, DMODEL * DMODEL);
    cvt_f32_f16<<<dim3(2048), blk, 0, stream>>>(W1, W1H, DMODEL * DFF);
    cvt_f32_f16<<<dim3(2048), blk, 0, stream>>>(W2, W2H, DFF * DMODEL);

    // QKV projections: [4096,1024] x [1024,1024] (+bias), f16 out
    gemm128_wmma<<<dim3(DMODEL / 128, BL / 128), blk, 0, stream>>>(
        Xh, DMODEL, WqP, DMODEL, nullptr, Qh, DMODEL, bq, 0, DMODEL);
    gemm128_wmma<<<dim3(DMODEL / 128, BL / 128), blk, 0, stream>>>(
        Xh, DMODEL, WkP, DMODEL, nullptr, Kh, DMODEL, bk, 0, DMODEL);
    gemm128_wmma<<<dim3(DMODEL / 128, BL / 128), blk, 0, stream>>>(
        Xh, DMODEL, WvP, DMODEL, nullptr, Vh, DMODEL, bv, 0, DMODEL);

    // attention (flash): 4096 wave-tiles, 8 waves/block
    flash_attn<<<dim3(BB * NH * (LL / 16) / 8), blk, 0, stream>>>(Qh, Kh, Vh, Oh);

    // output projection, f32 out
    gemm128_wmma<<<dim3(DMODEL / 128, BL / 128), blk, 0, stream>>>(
        Oh, DMODEL, WoH, DMODEL, mha, nullptr, DMODEL, nullptr, 0, DMODEL);

    // h1 = LN(x + mha)  (f32 + f16 copy)
    ln_res<<<dim3(BL), blk, 0, stream>>>(mha, x, ln1g, ln1b, h1f, h1h);

    // FFN up (+bias, relu), f16 out
    gemm128_wmma<<<dim3(DFF / 128, BL / 128), blk, 0, stream>>>(
        h1h, DMODEL, W1H, DFF, nullptr, ffa, DFF, b1, 1, DMODEL);
    // FFN down, f32 out (reuses mha buffer)
    gemm128_wmma<<<dim3(DMODEL / 128, BL / 128), blk, 0, stream>>>(
        ffa, DFF, W2H, DMODEL, ff, nullptr, DMODEL, nullptr, 0, DFF);

    // out = LN(h1 + ff)
    ln_res<<<dim3(BL), blk, 0, stream>>>(ff, h1f, ln2g, ln2b, out, nullptr);
}
